// FourierWeightAttention_12970801234300
// MI455X (gfx1250) — matrix operand order repaired and checked by reading, not compiled, not run on hardware
//
#include <hip/hip_runtime.h>
#include <stdint.h>
#include <stddef.h>

// ---------------------------------------------------------------------------
// FourierWeightAttention on gfx1250 (CDNA5).  All GEMM-shaped work runs on
// v_wmma_f32_16x16x32_bf16.  Attention is flash-style (online softmax), so the
// 512MB score tensor is never materialized; the 24MB bf16 Q/K/V working set
// lives in the 192MB L2.  GEMM K-loops are software-pipelined (fragment set
// k+1 loads issued before the wmmas of set k) so VMEM overlaps the matrix
// pipe instead of s_wait_loadcnt 0 before every wmma.
// ---------------------------------------------------------------------------

typedef __attribute__((ext_vector_type(16))) __bf16 v16bf;
typedef __attribute__((ext_vector_type(8)))  float  v8f;
typedef __attribute__((ext_vector_type(4)))  float  f32x4;
typedef __attribute__((ext_vector_type(4)))  __bf16 bf16x4;

#define D_  512
#define H_  8
#define HD_ 64
#define B_  4
#define S_  2048
#define M_  (B_ * S_)                 // 8192 rows of x
#define SCALE_ 0.125f                 // 1/sqrt(64)
#define PROJ_SCALE_ 0.04419417382415922f  // sqrt(2/(2*512))

// ---- WMMA helpers ---------------------------------------------------------

__device__ __forceinline__ v8f wmma_bf16(v16bf a, v16bf b, v8f c) {
  // (neg_a, A, neg_b, B, c_mod, C, reuse_a, reuse_b)
  return __builtin_amdgcn_wmma_f32_16x16x32_bf16(false, a, false, b,
                                                 (short)0, c, false, false);
}

union Frag16 { v16bf v; f32x4 f[2]; };

// B-matrix fragment (32xK-contig x 16): lane takes k = {8h..8h+7, 16+8h..16+8h+7}
// of a 32-element contraction chunk.
__device__ __forceinline__ v16bf load_fragB(const __bf16* chunk, int half) {
  Frag16 u;
  u.f[0] = *(const f32x4*)(chunk + half * 8);
  u.f[1] = *(const f32x4*)(chunk + 16 + half * 8);
  return u.v;
}

// A-matrix fragment (16x32 row-major): lane's row, k = {8h..8h+7, 16+8h..16+8h+7}
__device__ __forceinline__ v16bf load_fragA(const __bf16* row, int half) {
  Frag16 u;
  u.f[0] = *(const f32x4*)(row + half * 8);
  u.f[1] = *(const f32x4*)(row + 16 + half * 8);
  return u.v;
}

__device__ __forceinline__ v8f v8f_zero() {
  v8f z = {0.f, 0.f, 0.f, 0.f, 0.f, 0.f, 0.f, 0.f};
  return z;
}

// ---- 1) fp32 -> bf16 conversion (x, W_out) --------------------------------

__global__ void f32_to_bf16_vec4(const float* __restrict__ src,
                                 __bf16* __restrict__ dst, int n4) {
  int i = blockIdx.x * blockDim.x + threadIdx.x;
  if (i >= n4) return;
  f32x4 v = ((const f32x4*)src)[i];
  bf16x4 o;
  o.x = (__bf16)v.x; o.y = (__bf16)v.y; o.z = (__bf16)v.z; o.w = (__bf16)v.w;
  ((bf16x4*)dst)[i] = o;
}

// ---- 2) Fourier weight generation: W = (features @ alpha) * PROJ_SCALE ----
// features: (262144, 128) fp32, streamed once (134MB -> pure bandwidth,
// ~5.8us at 23.3 TB/s).  One thread per output element, float4 streaming;
// alpha vectors are wave-uniform -> scalar cache.

__global__ void gen_weights(const float* __restrict__ feat,
                            const float* __restrict__ aq,
                            const float* __restrict__ ak,
                            const float* __restrict__ av,
                            __bf16* __restrict__ Wq,
                            __bf16* __restrict__ Wk,
                            __bf16* __restrict__ Wv) {
  int j = blockIdx.x * blockDim.x + threadIdx.x;        // 0..262143
  const f32x4* f = (const f32x4*)(feat + (size_t)j * 128);
  const f32x4* A = (const f32x4*)aq;
  const f32x4* Bp = (const f32x4*)ak;
  const f32x4* C = (const f32x4*)av;
  float sq = 0.f, sk = 0.f, sv = 0.f;
#pragma unroll 8
  for (int i = 0; i < 32; ++i) {
    f32x4 fv = f[i];
    f32x4 a = A[i], b = Bp[i], c = C[i];
    sq += fv.x * a.x + fv.y * a.y + fv.z * a.z + fv.w * a.w;
    sk += fv.x * b.x + fv.y * b.y + fv.z * b.z + fv.w * b.w;
    sv += fv.x * c.x + fv.y * c.y + fv.z * c.z + fv.w * c.w;
  }
  Wq[j] = (__bf16)(sq * PROJ_SCALE_);
  Wk[j] = (__bf16)(sk * PROJ_SCALE_);
  Wv[j] = (__bf16)(sv * PROJ_SCALE_);
}

// ---- 3) QKV projection:  y = x @ W^T + bias -------------------------------
// One wave computes a 16(M) x 64(N) strip for one head.  K-loop = 512/32 = 16
// steps, 4 WMMA per step, software-pipelined (next A + 4 B fragments in
// flight while current 4 wmmas execute).  Q is pre-scaled by 1/sqrt(hd);
// V is written TRANSPOSED [b][h][hd][s] so the attention PV GEMM gets a
// contraction-contiguous B operand.

__global__ void qkv_proj(const __bf16* __restrict__ xb,
                         const __bf16* __restrict__ Wq,
                         const __bf16* __restrict__ Wk,
                         const __bf16* __restrict__ Wv,
                         const float* __restrict__ bq,
                         const float* __restrict__ bk,
                         const float* __restrict__ bv,
                         __bf16* __restrict__ Qb,
                         __bf16* __restrict__ Kb,
                         __bf16* __restrict__ Vt) {
  const int lane = threadIdx.x & 31;
  const int wid  = threadIdx.x >> 5;
  const int half = lane >> 4;
  const int lm   = lane & 15;
  const int m0   = (blockIdx.x * 4 + wid) * 16;   // row tile base
  const int h    = blockIdx.y;                    // head
  const int nbase = h * HD_;
  const int which = blockIdx.z;                   // 0=Q 1=K 2=V

  const __bf16* W    = (which == 0) ? Wq : (which == 1) ? Wk : Wv;
  const float*  bias = (which == 0) ? bq : (which == 1) ? bk : bv;

  v8f acc[4];
#pragma unroll
  for (int t = 0; t < 4; ++t) acc[t] = v8f_zero();

  const __bf16* xrow = xb + (size_t)(m0 + lm) * D_;
  const __bf16* wrow[4];
#pragma unroll
  for (int t = 0; t < 4; ++t) wrow[t] = W + (size_t)(nbase + t * 16 + lm) * D_;

  // software pipeline: fragments for step i+1 are loaded before step i's wmmas
  v16bf a0 = load_fragA(xrow, half);
  v16bf b0[4];
#pragma unroll
  for (int t = 0; t < 4; ++t) b0[t] = load_fragB(wrow[t], half);

#pragma unroll
  for (int i = 0; i < 15; ++i) {
    const int kc = (i + 1) * 32;
    v16bf a1 = load_fragA(xrow + kc, half);
    v16bf b1[4];
#pragma unroll
    for (int t = 0; t < 4; ++t) b1[t] = load_fragB(wrow[t] + kc, half);
#pragma unroll
    for (int t = 0; t < 4; ++t) acc[t] = wmma_bf16(a0, b0[t], acc[t]);
    a0 = a1;
#pragma unroll
    for (int t = 0; t < 4; ++t) b0[t] = b1[t];
  }
#pragma unroll
  for (int t = 0; t < 4; ++t) acc[t] = wmma_bf16(a0, b0[t], acc[t]);

  const int bidx  = m0 >> 11;                     // batch (S_=2048)
  const int sbase = (m0 & (S_ - 1)) + half * 8;   // first seq row this lane owns
  const int bh    = bidx * H_ + h;

  if (which == 2) {
    // V transposed: [bh][hd][s]; 8 consecutive s per lane -> one b128 store
#pragma unroll
    for (int t = 0; t < 4; ++t) {
      int hd = t * 16 + lm;
      float bvl = bias[nbase + hd];
      union { f32x4 v; __bf16 hh[8]; } pk;
#pragma unroll
      for (int r = 0; r < 8; ++r) pk.hh[r] = (__bf16)(acc[t][r] + bvl);
      *(f32x4*)(Vt + ((size_t)bh * HD_ + hd) * S_ + sbase) = pk.v;
    }
  } else {
    __bf16* dst = (which == 0) ? Qb : Kb;
    const float scl = (which == 0) ? SCALE_ : 1.0f;
#pragma unroll
    for (int t = 0; t < 4; ++t) {
      int hd = t * 16 + lm;
      float bvl = bias[nbase + hd];
#pragma unroll
      for (int r = 0; r < 8; ++r) {
        int s = sbase + r;
        dst[((size_t)bh * S_ + s) * HD_ + hd] = (__bf16)((acc[t][r] + bvl) * scl);
      }
    }
  }
}

// ---- 4) Flash attention ---------------------------------------------------
// One wave owns a 16-query tile for one (b,h).  Per 32-key block:
//   S = Q(16x64) @ K^T          -> 4 WMMA (K frags preloaded last iteration)
//   issue V frag loads + next-block K frag loads (hidden under softmax VALU)
//   online softmax (shuffle-reduce over the 16-lane half that owns a row)
//   P C-frag -> LDS -> A-frag re-layout (1KB per wave)
//   O(16x64) += P(16x32) @ V    -> 4 WMMA (B from transposed V)

__global__ void flash_attn(const __bf16* __restrict__ Qb,
                           const __bf16* __restrict__ Kb,
                           const __bf16* __restrict__ Vt,
                           __bf16* __restrict__ Ob) {
  __shared__ __align__(16) __bf16 Pl[4][16 * 32];

  const int lane = threadIdx.x & 31;
  const int wid  = threadIdx.x >> 5;
  const int half = lane >> 4;
  const int lm   = lane & 15;
  const int q0   = (blockIdx.x * 4 + wid) * 16;
  const int bh   = blockIdx.y;

  const __bf16* Qp = Qb + (size_t)bh * S_ * HD_;
  const __bf16* Kp = Kb + (size_t)bh * S_ * HD_;
  const __bf16* Vp = Vt + (size_t)bh * HD_ * S_;
  __bf16* P = Pl[wid];

  // Q A-frags (k-dim 64 -> 2 chunks), resident for the whole key loop
  const __bf16* qrow = Qp + (size_t)(q0 + lm) * HD_;
  v16bf qa0 = load_fragA(qrow, half);
  v16bf qa1 = load_fragA(qrow + 32, half);

  float mr[8], lr[8], cr[8];
  v8f o[4];
#pragma unroll
  for (int r = 0; r < 8; ++r) { mr[r] = -1e30f; lr[r] = 0.f; }
#pragma unroll
  for (int t = 0; t < 4; ++t) o[t] = v8f_zero();

  // preload the first key block's K fragments
  v16bf kb[4];
  kb[0] = load_fragB(Kp + (size_t)(lm) * HD_, half);
  kb[1] = load_fragB(Kp + (size_t)(16 + lm) * HD_, half);
  kb[2] = load_fragB(Kp + (size_t)(lm) * HD_ + 32, half);
  kb[3] = load_fragB(Kp + (size_t)(16 + lm) * HD_ + 32, half);

  for (int k0 = 0; k0 < S_; k0 += 32) {
    // scores for this block (consume preloaded kb)
    v8f s0 = v8f_zero(), s1 = v8f_zero();
    s0 = wmma_bf16(qa0, kb[0], s0);
    s1 = wmma_bf16(qa0, kb[1], s1);
    s0 = wmma_bf16(qa1, kb[2], s0);
    s1 = wmma_bf16(qa1, kb[3], s1);

    // this block's V fragments -- consumed after softmax+LDS, loads overlap
    v16bf vb[4];
#pragma unroll
    for (int t = 0; t < 4; ++t)
      vb[t] = load_fragB(Vp + (size_t)(t * 16 + lm) * S_ + k0, half);

    // next block's K fragments -- hidden under the softmax VALU below
    const int kn = k0 + 32;
    if (kn < S_) {
      kb[0] = load_fragB(Kp + (size_t)(kn + lm) * HD_, half);
      kb[1] = load_fragB(Kp + (size_t)(kn + 16 + lm) * HD_, half);
      kb[2] = load_fragB(Kp + (size_t)(kn + lm) * HD_ + 32, half);
      kb[3] = load_fragB(Kp + (size_t)(kn + 16 + lm) * HD_ + 32, half);
    }
    // warm L2/WGP$ two blocks ahead (global_prefetch_b8)
    if (kn + 32 < S_) {
      __builtin_prefetch(Kp + (size_t)(kn + 32 + lm) * HD_, 0, 1);
      __builtin_prefetch(Vp + (size_t)lm * S_ + kn + 32, 0, 1);
      __builtin_prefetch(Vp + (size_t)(32 + lm) * S_ + kn + 32, 0, 1);
    }

    // online softmax over this 32-key block; each row lives in one 16-lane half
#pragma unroll
    for (int r = 0; r < 8; ++r) {
      float mx = fmaxf(s0[r], s1[r]);
      mx = fmaxf(mx, __shfl_xor(mx, 1));
      mx = fmaxf(mx, __shfl_xor(mx, 2));
      mx = fmaxf(mx, __shfl_xor(mx, 4));
      mx = fmaxf(mx, __shfl_xor(mx, 8));
      float mnew = fmaxf(mr[r], mx);
      float p0 = __expf(s0[r] - mnew);
      float p1 = __expf(s1[r] - mnew);
      float rs = p0 + p1;
      rs += __shfl_xor(rs, 1);
      rs += __shfl_xor(rs, 2);
      rs += __shfl_xor(rs, 4);
      rs += __shfl_xor(rs, 8);
      float corr = __expf(mr[r] - mnew);
      lr[r] = lr[r] * corr + rs;
      mr[r] = mnew;
      cr[r] = corr;
      int row = half * 8 + r;               // C-frag row this lane holds
      P[row * 32 + lm]      = (__bf16)p0;
      P[row * 32 + 16 + lm] = (__bf16)p1;
    }
#pragma unroll
    for (int t = 0; t < 4; ++t)
#pragma unroll
      for (int r = 0; r < 8; ++r) o[t][r] *= cr[r];

    __syncthreads();                         // ds stores visible before loads
    v16bf pa = load_fragA(P + lm * 32, half);
#pragma unroll
    for (int t = 0; t < 4; ++t) o[t] = wmma_bf16(pa, vb[t], o[t]);
    __syncthreads();                         // before next block rewrites P
  }

  const int b = bh >> 3, h = bh & 7;
#pragma unroll
  for (int r = 0; r < 8; ++r) cr[r] = 1.0f / lr[r];
#pragma unroll
  for (int t = 0; t < 4; ++t) {
    int n = h * HD_ + t * 16 + lm;
#pragma unroll
    for (int r = 0; r < 8; ++r) {
      int m = b * S_ + q0 + half * 8 + r;
      Ob[(size_t)m * D_ + n] = (__bf16)(o[t][r] * cr[r]);
    }
  }
}

// ---- 5) Output projection:  out = O @ W_out^T + b_out  (fp32 out) ---------

__global__ void out_proj(const __bf16* __restrict__ Ob,
                         const __bf16* __restrict__ Wo,
                         const float* __restrict__ bo,
                         float* __restrict__ out) {
  const int lane = threadIdx.x & 31;
  const int wid  = threadIdx.x >> 5;
  const int half = lane >> 4;
  const int lm   = lane & 15;
  const int m0   = (blockIdx.x * 4 + wid) * 16;
  const int nbase = blockIdx.y * 64;

  v8f acc[4];
#pragma unroll
  for (int t = 0; t < 4; ++t) acc[t] = v8f_zero();

  const __bf16* orow = Ob + (size_t)(m0 + lm) * D_;
  const __bf16* wrow[4];
#pragma unroll
  for (int t = 0; t < 4; ++t) wrow[t] = Wo + (size_t)(nbase + t * 16 + lm) * D_;

  v16bf a0 = load_fragA(orow, half);
  v16bf b0[4];
#pragma unroll
  for (int t = 0; t < 4; ++t) b0[t] = load_fragB(wrow[t], half);

#pragma unroll
  for (int i = 0; i < 15; ++i) {
    const int kc = (i + 1) * 32;
    v16bf a1 = load_fragA(orow + kc, half);
    v16bf b1[4];
#pragma unroll
    for (int t = 0; t < 4; ++t) b1[t] = load_fragB(wrow[t] + kc, half);
#pragma unroll
    for (int t = 0; t < 4; ++t) acc[t] = wmma_bf16(a0, b0[t], acc[t]);
    a0 = a1;
#pragma unroll
    for (int t = 0; t < 4; ++t) b0[t] = b1[t];
  }
#pragma unroll
  for (int t = 0; t < 4; ++t) acc[t] = wmma_bf16(a0, b0[t], acc[t]);

#pragma unroll
  for (int t = 0; t < 4; ++t) {
    int n = nbase + t * 16 + lm;
    float bvl = bo[n];
#pragma unroll
    for (int r = 0; r < 8; ++r) {
      int m = m0 + half * 8 + r;
      out[(size_t)m * D_ + n] = acc[t][r] + bvl;
    }
  }
}

// ---- launch ---------------------------------------------------------------

extern "C" void kernel_launch(void* const* d_in, const int* in_sizes, int n_in,
                              void* d_out, int out_size, void* d_ws, size_t ws_size,
                              hipStream_t stream) {
  (void)in_sizes; (void)n_in; (void)out_size;

  const float* x    = (const float*)d_in[0];
  const float* aq   = (const float*)d_in[1];
  const float* ak   = (const float*)d_in[2];
  const float* av   = (const float*)d_in[3];
  const float* bq   = (const float*)d_in[4];
  const float* bk   = (const float*)d_in[5];
  const float* bv   = (const float*)d_in[6];
  const float* Wout = (const float*)d_in[7];
  const float* bo   = (const float*)d_in[8];
  const float* feat = (const float*)d_in[9];
  float* out = (float*)d_out;

  // workspace carve (bytes)
  const size_t SZ_W  = (size_t)D_ * D_ * 2;          // 512KB per bf16 weight
  const size_t SZ_XB = (size_t)M_ * D_ * 2;          // 8MB per bf16 activation
  char* ws = (char*)d_ws;
  size_t off = 0;
  __bf16* Wq = (__bf16*)(ws + off); off += SZ_W;
  __bf16* Wk = (__bf16*)(ws + off); off += SZ_W;
  __bf16* Wv = (__bf16*)(ws + off); off += SZ_W;
  __bf16* Wo = (__bf16*)(ws + off); off += SZ_W;
  __bf16* xb = (__bf16*)(ws + off); off += SZ_XB;
  __bf16* Qb = (__bf16*)(ws + off); off += SZ_XB;
  __bf16* Kb = (__bf16*)(ws + off); off += SZ_XB;
  __bf16* Vt = (__bf16*)(ws + off); off += SZ_XB;
  __bf16* Ob = (__bf16*)(ws + off); off += SZ_XB;
  if (ws_size < off) return;                          // ~42MB needed

  // 1) bf16 conversions
  {
    int n4 = (M_ * D_) / 4;                           // x: 4,194,304 elems
    f32_to_bf16_vec4<<<dim3(n4 / 256), dim3(256), 0, stream>>>(x, xb, n4);
  }
  {
    int n4 = (D_ * D_) / 4;                           // W_out: 262,144 elems
    f32_to_bf16_vec4<<<dim3(n4 / 256), dim3(256), 0, stream>>>(Wout, Wo, n4);
  }

  // 2) Fourier weights (bandwidth-bound: streams 134MB of features once)
  gen_weights<<<dim3((D_ * D_) / 256), dim3(256), 0, stream>>>(
      feat, aq, ak, av, Wq, Wk, Wv);

  // 3) QKV projection: grid = (M/16/4 mtiles, heads, {q,k,v}), 4 waves/block
  qkv_proj<<<dim3(M_ / 64, H_, 3), dim3(128), 0, stream>>>(
      xb, Wq, Wk, Wv, bq, bk, bv, Qb, Kb, Vt);

  // 4) Flash attention: grid = (qtiles/4, B*H), 4 waves/block
  flash_attn<<<dim3(S_ / 64, B_ * H_), dim3(128), 0, stream>>>(Qb, Kb, Vt, Ob);

  // 5) Output projection
  out_proj<<<dim3(M_ / 64, D_ / 64), dim3(128), 0, stream>>>(Ob, Wo, bo, out);
}
